// NetworkWithAttention_48086453846758
// MI455X (gfx1250) — matrix operand order repaired and checked by reading, not compiled, not run on hardware
//
#include <hip/hip_runtime.h>

// ============================================================================
// MI455X / gfx1250 implementation.
//
// Roofline: 59 GFLOP total, ~55 GFLOP dense GEMM -> v_wmma_f32_16x16x32_f16
// (f16 in, f32 accumulate). Activation buffer [B,180] = 94 MB lives in the
// 192 MB L2 across the 8 BatchNorm-forced kernel boundaries, so the ~10
// full passes over it run at L2 bandwidth; only x (127 MB) and the output
// (5 MB) must stream from HBM (23.3 TB/s -> ~6 us floor).
// Data movement: each workgroup's contiguous 46 KB activation tile moves
// global<->LDS via the Tensor Data Mover (tensor_load_to_lds /
// tensor_store_from_lds, TENSORcnt), not per-lane VMEM, so the VALU only
// does BN + f32->f16 conversion and WMMA fragment packing.
// BatchNorm statistics: fp32, deterministic (per-block partials written by
// block index, then a fixed-order 180-thread reduction; no float atomics).
// ============================================================================

namespace {
constexpr int   kB    = 131072;
constexpr int   kS    = 9;
constexpr int   kD    = 20;
constexpr int   kH    = 100;
constexpr int   kSD   = kS * kD;     // 180
constexpr int   kXD   = 27;
constexpr int   kSPB  = 64;          // samples per stage block (4 waves x 16)
constexpr int   kNBLK = kB / kSPB;   // 2048
constexpr float kEps  = 1e-5f;
}

typedef __attribute__((ext_vector_type(16))) _Float16     v16h;
typedef __attribute__((ext_vector_type(8)))  float        v8f;
typedef __attribute__((ext_vector_type(4)))  unsigned int u32x4;
typedef __attribute__((ext_vector_type(8)))  int          i32x8;
typedef __attribute__((ext_vector_type(4)))  int          i32x4;

// CDNA5 WMMA 16x16x32 f16 VGPR layout helpers (ISA 7.12.2, wave32):
// A (16x32, 16-bit): row M = lane&15; lanes 0-15 hold K 0-7 / 16-23 in the
//   16 halves, lanes 16-31 hold K 8-15 / 24-31.
__device__ __forceinline__ int a_k(int e, int lane) {
  return ((e < 8) ? e : e + 8) + ((lane & 16) ? 8 : 0);
}
// B (32x16, 16-bit): N = lane&15; same K order as A (lanes 0-15: K 0-7 / 16-23,
//   lanes 16-31: K 8-15 / 24-31).
__device__ __forceinline__ int b_k(int e, int lane) {
  return ((e < 8) ? e : e + 8) + ((lane & 16) ? 8 : 0);
}
// C/D (16x16 f32): N = lane&15; element e -> M = e + 8*(lane>=16).
__device__ __forceinline__ int d_m(int e, int lane) {
  return e + ((lane & 16) ? 8 : 0);
}
__device__ __forceinline__ v8f vzero8() {
  v8f z = {0.f, 0.f, 0.f, 0.f, 0.f, 0.f, 0.f, 0.f};
  return z;
}
__device__ __forceinline__ v8f wmma16(v16h a, v16h b, v8f c) {
  return __builtin_amdgcn_wmma_f32_16x16x32_f16(false, a, false, b,
                                                (short)0, c, false, false);
}

// ---------------------------------------------------------------------------
// Tensor Data Mover: 1-D contiguous f32 tile, global<->LDS.
// D# per ISA ch.8: group0 = {count=1 | lds_addr | global_addr(57) | type=2},
// group1 = {data_size=4B, tensor_dim0=n, tile_dim0=n, tile_dim1=1,
//           dim0_stride=n}; groups 2/3 unused (tile_dim2=0).
// Issue from one wave only (EXEC ignored; one TDM op per instruction).
// Toolchain uses the 6-operand builtin form (extra descriptor group unused
// for <=2-D tiles; zero-filled).
// ---------------------------------------------------------------------------
__device__ __forceinline__ void tdm_desc_1d(u32x4& g0, i32x8& g1,
                                            unsigned lds_off,
                                            unsigned long long gaddr, int n) {
  g0[0] = 1u;                                    // count=1, user descriptor
  g0[1] = lds_off;                               // LDS byte address
  g0[2] = (unsigned)(gaddr & 0xffffffffull);     // global addr [31:0]
  g0[3] = (unsigned)((gaddr >> 32) & 0x1ffffffull) | (2u << 30);  // type=2
  g1[0] = 2 << 16;                               // data_size = 4 bytes
  g1[1] = (n & 0xffff) << 16;                    // tensor_dim0 [15:0]
  g1[2] = ((n >> 16) & 0xffff) | (1 << 16);      // tensor_dim0 hi, tensor_dim1=1
  g1[3] = (n & 0xffff) << 16;                    // tensor_dim1 hi=0, tile_dim0=n
  g1[4] = 1;                                     // tile_dim1=1, tile_dim2=0
  g1[5] = n;                                     // tensor_dim0_stride lo32
  g1[6] = 0;
  g1[7] = 0;
}
__device__ __forceinline__ void tdm_load_1d(void* lds, const void* gptr, int n) {
  u32x4 g0; i32x8 g1;
  i32x4 gz4 = {0, 0, 0, 0};
  i32x8 gz8 = {0, 0, 0, 0, 0, 0, 0, 0};
  tdm_desc_1d(g0, g1, (unsigned)(unsigned long long)lds,
              (unsigned long long)gptr, n);
  __builtin_amdgcn_tensor_load_to_lds(g0, g1, gz4, gz4, gz8, 0);
}
__device__ __forceinline__ void tdm_store_1d(void* lds, void* gptr, int n) {
  u32x4 g0; i32x8 g1;
  i32x4 gz4 = {0, 0, 0, 0};
  i32x8 gz8 = {0, 0, 0, 0, 0, 0, 0, 0};
  tdm_desc_1d(g0, g1, (unsigned)(unsigned long long)lds,
              (unsigned long long)gptr, n);
  __builtin_amdgcn_tensor_store_from_lds(g0, g1, gz4, gz4, gz8, 0);
}

// ---------------------------------------------------------------------------
// Embed: h[(b,s), d] = sum_i x[(b,s), i] * ew[d, i].  GEMM (B*S)x27 * 27x20.
// 256 threads = 8 waves, 16 rows/wave -> 128 rows/block.  x tile arrives in
// LDS via TDM (this is the one true HBM stream).
// ---------------------------------------------------------------------------
__global__ __launch_bounds__(256) void k_embed(const float* __restrict__ x,
                                               const float* __restrict__ ew,
                                               float* __restrict__ hbuf) {
  __shared__ float    lraw[128 * kXD];  // raw x tile (TDM landing)
  __shared__ _Float16 lx[128 * 28];     // f16 x tile, padded col 27 = 0
  __shared__ float    lw[kD * kXD];     // 540
  const int  tid = threadIdx.x;
  const long r0  = (long)blockIdx.x * 128;

  if (tid < 32) {                       // wave 0 drives the DMA
    tdm_load_1d(lraw, x + r0 * kXD, 128 * kXD);
    __builtin_amdgcn_s_wait_tensorcnt(0);
  }
  for (int t = tid; t < kD * kXD; t += 256) lw[t] = ew[t];
  __syncthreads();

  for (int t = tid; t < 128 * kXD; t += 256) {
    int r = t / kXD, c = t % kXD;
    lx[r * 28 + c] = (_Float16)lraw[t];
  }
  for (int t = tid; t < 128; t += 256) lx[t * 28 + 27] = (_Float16)0.f;
  __syncthreads();

  const int lane = tid & 31, wv = tid >> 5;
  const int arow = wv * 16 + (lane & 15);
  v16h a;
#pragma unroll
  for (int e = 0; e < 16; ++e) {
    int k = a_k(e, lane);                       // K pad 27->32; col27 is 0
    a[e] = (k < 28) ? lx[arow * 28 + k] : (_Float16)0.f;
  }
  v16h b0, b1;
#pragma unroll
  for (int e = 0; e < 16; ++e) {
    int k  = b_k(e, lane);
    int n  = lane & 15;
    int n1 = n + 16;
    b0[e] = (k < kXD) ? (_Float16)lw[n * kXD + k] : (_Float16)0.f;
    b1[e] = (k < kXD && n1 < kD) ? (_Float16)lw[n1 * kXD + k] : (_Float16)0.f;
  }
  v8f c0 = wmma16(a, b0, vzero8());
  v8f c1 = wmma16(a, b1, vzero8());
#pragma unroll
  for (int e = 0; e < 8; ++e) {
    int  m   = d_m(e, lane);
    long row = r0 + wv * 16 + m;
    int  n   = lane & 15;
    hbuf[row * kD + n] = c0[e];                 // [B*S][20] == [B][180]
    if (n + 16 < kD) hbuf[row * kD + n + 16] = c1[e];
  }
}

// ---------------------------------------------------------------------------
// Attention stage: (optional BN of previous stage) -> h += attn(h); writes
// per-block sum/sumsq partials for its BN stage.
// 128 threads = 4 waves, 16 samples/wave.  Tile in/out via TDM; QKV on WMMA
// (M = 16 samples per token); 9x9 scores/softmax/AV scalar (~6% of FLOPs).
// ---------------------------------------------------------------------------
__global__ __launch_bounds__(128) void k_attn(float* __restrict__ hbuf,
                                              const float* __restrict__ w,
                                              const float* __restrict__ bias,
                                              const float* __restrict__ bnsc,
                                              float* __restrict__ pS,
                                              float* __restrict__ pQ) {
  __shared__ float    lhn[kSPB * kSD];              // TDM tile / f32 running h
  __shared__ _Float16 lh[kSPB * kSD];               // post-BN h (f16, WMMA A)
  __shared__ _Float16 lqkv[3 * kSPB * kS * kD];     // q,k,v (f16)
  __shared__ float    lw[3 * kD * kD];
  __shared__ float    lb[3 * kD];

  const int  tid = threadIdx.x;
  const long s0  = (long)blockIdx.x * kSPB;

  if (tid < 32) {
    tdm_load_1d(lhn, hbuf + s0 * kSD, kSPB * kSD);
    __builtin_amdgcn_s_wait_tensorcnt(0);
  }
  for (int t = tid; t < 3 * kD * kD; t += 128) lw[t] = w[t];
  for (int t = tid; t < 3 * kD; t += 128) lb[t] = bias[t];
  __syncthreads();

  // BN (in place) + f16 copy for the WMMA A operands.
  for (int t = tid; t < kSPB * kSD; t += 128) {
    float v = lhn[t];
    if (bnsc) {
      int f = t % kSD;
      v = v * bnsc[f] + bnsc[kSD + f];
      lhn[t] = v;
    }
    lh[t] = (_Float16)v;
  }
  __syncthreads();

  const int lane = tid & 31, wv = tid >> 5;
  // 6 B fragments (q/k/v x 2 N-tiles), resident in VGPRs for all 9 tokens.
  v16h bf[3][2];
#pragma unroll
  for (int p = 0; p < 3; ++p)
#pragma unroll
    for (int nt = 0; nt < 2; ++nt)
#pragma unroll
      for (int e = 0; e < 16; ++e) {
        int k = b_k(e, lane);
        int n = nt * 16 + (lane & 15);
        bf[p][nt][e] = (k < kD && n < kD) ? (_Float16)lw[p * kD * kD + n * kD + k]
                                          : (_Float16)0.f;
      }
  for (int s = 0; s < kS; ++s) {
    const int sl = wv * 16 + (lane & 15);
    v16h a;
#pragma unroll
    for (int e = 0; e < 16; ++e) {
      int k = a_k(e, lane);
      a[e] = (k < kD) ? lh[sl * kSD + s * kD + k] : (_Float16)0.f;
    }
#pragma unroll
    for (int p = 0; p < 3; ++p)
#pragma unroll
      for (int nt = 0; nt < 2; ++nt) {
        v8f c = wmma16(a, bf[p][nt], vzero8());
#pragma unroll
        for (int e = 0; e < 8; ++e) {
          int n = nt * 16 + (lane & 15);
          if (n < kD) {
            int smp = wv * 16 + d_m(e, lane);
            lqkv[((p * kSPB + smp) * kS + s) * kD + n] =
                (_Float16)(c[e] + lb[p * kD + n]);
          }
        }
      }
  }
  __syncthreads();

  // scores + softmax + A*V: 2 threads per sample, alternate token rows.
  const float rs = 0.22360679774997896f;  // 1/sqrt(20)
  {
    const int sl = tid >> 1;
    for (int s = (tid & 1); s < kS; s += 2) {
      float q[kD];
#pragma unroll
      for (int d = 0; d < kD; ++d)
        q[d] = (float)lqkv[((0 * kSPB + sl) * kS + s) * kD + d];
      float sc[kS];
      float mx = -1e30f;
      for (int t2 = 0; t2 < kS; ++t2) {
        float acc = 0.f;
#pragma unroll
        for (int d = 0; d < kD; ++d)
          acc += q[d] * (float)lqkv[((1 * kSPB + sl) * kS + t2) * kD + d];
        acc *= rs;
        sc[t2] = acc;
        mx = fmaxf(mx, acc);
      }
      float sum = 0.f;
      for (int t2 = 0; t2 < kS; ++t2) {
        sc[t2] = __expf(sc[t2] - mx);
        sum += sc[t2];
      }
      const float inv = 1.f / sum;
#pragma unroll
      for (int d = 0; d < kD; ++d) {
        float acc = 0.f;
        for (int t2 = 0; t2 < kS; ++t2)
          acc += sc[t2] * (float)lqkv[((2 * kSPB + sl) * kS + t2) * kD + d];
        lhn[sl * kSD + s * kD + d] += acc * inv;   // residual
      }
    }
  }
  __syncthreads();

  if (tid < 32) tdm_store_1d(lhn, hbuf + s0 * kSD, kSPB * kSD);  // DMA back
  for (int f = tid; f < kSD; f += 128) {
    float s1 = 0.f, s2 = 0.f;
    for (int sl = 0; sl < kSPB; ++sl) {
      float v = lhn[sl * kSD + f];
      s1 += v;
      s2 += v * v;
    }
    pS[(long)blockIdx.x * kSD + f] = s1;
    pQ[(long)blockIdx.x * kSD + f] = s2;
  }
  // S_ENDPGM performs an implicit wait-idle (covers TENSORcnt).
}

// ---------------------------------------------------------------------------
// BN stat reduction: fixed-order over 2048 block partials -> scale/shift.
// ---------------------------------------------------------------------------
__global__ __launch_bounds__(192) void k_red(const float* __restrict__ pS,
                                             const float* __restrict__ pQ,
                                             const float* __restrict__ gamma,
                                             const float* __restrict__ beta,
                                             float* __restrict__ bnsc,
                                             int nblk) {
  const int f = threadIdx.x;
  if (f >= kSD) return;
  float s1 = 0.f, s2 = 0.f;
  for (int b = 0; b < nblk; ++b) {
    s1 += pS[(long)b * kSD + f];
    s2 += pQ[(long)b * kSD + f];
  }
  const float invB = 1.f / (float)kB;
  const float mean = s1 * invB;
  const float var  = s2 * invB - mean * mean;
  const float sc   = gamma[f] * rsqrtf(var + kEps);
  bnsc[f]        = sc;
  bnsc[kSD + f]  = beta[f] - mean * sc;
}

// ---------------------------------------------------------------------------
// MLP stage: h = BN(h); h += relu(relu(h*W1^T+b1)*W2^T+b2).  Both layers on
// WMMA; rows = (sample, token), 9 M-tiles per wave, layer1->layer2 handoff
// through a per-wave 16x112 LDS transpose stage (D-layout -> A-layout).
// ---------------------------------------------------------------------------
__global__ __launch_bounds__(128) void k_mlp(float* __restrict__ hbuf,
                                             const float* __restrict__ w1,
                                             const float* __restrict__ b1,
                                             const float* __restrict__ w2,
                                             const float* __restrict__ b2,
                                             const float* __restrict__ bnsc,
                                             float* __restrict__ pS,
                                             float* __restrict__ pQ) {
  __shared__ float    lhn[kSPB * kSD];  // TDM tile / f32 running h
  __shared__ _Float16 lh[kSPB * kSD];   // post-BN h (f16)
  __shared__ _Float16 lt[4][16 * 112];  // per-wave relu(layer1) staging
  __shared__ float    lw1[kH * kD];
  __shared__ float    lb1[kH];
  __shared__ float    lw2[kD * kH];
  __shared__ float    lb2[kD];

  const int  tid = threadIdx.x;
  const long s0  = (long)blockIdx.x * kSPB;

  if (tid < 32) {
    tdm_load_1d(lhn, hbuf + s0 * kSD, kSPB * kSD);
    __builtin_amdgcn_s_wait_tensorcnt(0);
  }
  for (int t = tid; t < kH * kD; t += 128) lw1[t] = w1[t];
  for (int t = tid; t < kH; t += 128) lb1[t] = b1[t];
  for (int t = tid; t < kD * kH; t += 128) lw2[t] = w2[t];
  for (int t = tid; t < kD; t += 128) lb2[t] = b2[t];
  __syncthreads();

  for (int t = tid; t < kSPB * kSD; t += 128) {
    int f = t % kSD;
    float v = lhn[t] * bnsc[f] + bnsc[kSD + f];
    lhn[t] = v;
    lh[t]  = (_Float16)v;
  }
  __syncthreads();

  const int lane = tid & 31, wv = tid >> 5;
  // Layer1 B fragments: K=20 (pad 32), N=100 (pad 112 -> 7 N-tiles).
  v16h bw1[7];
#pragma unroll
  for (int n = 0; n < 7; ++n)
#pragma unroll
    for (int e = 0; e < 16; ++e) {
      int k   = b_k(e, lane);
      int col = n * 16 + (lane & 15);
      bw1[n][e] = (k < kD && col < kH) ? (_Float16)lw1[col * kD + k]
                                       : (_Float16)0.f;
    }
  // Layer2 B fragments: K=100 (4 x 32), N=20 (2 N-tiles).
  v16h bw2[4][2];
#pragma unroll
  for (int kk = 0; kk < 4; ++kk)
#pragma unroll
    for (int nt = 0; nt < 2; ++nt)
#pragma unroll
      for (int e = 0; e < 16; ++e) {
        int k = kk * 32 + b_k(e, lane);
        int n = nt * 16 + (lane & 15);
        bw2[kk][nt][e] = (k < kH && n < kD) ? (_Float16)lw2[n * kH + k]
                                            : (_Float16)0.f;
      }

  for (int mt = 0; mt < 9; ++mt) {        // 144 rows per wave / 16
    const int row = wv * 144 + mt * 16 + (lane & 15);  // = sl*9 + s (local)
    const int sl  = row / kS, s = row % kS;
    v16h a;
#pragma unroll
    for (int e = 0; e < 16; ++e) {
      int k = a_k(e, lane);
      a[e] = (k < kD) ? lh[sl * kSD + s * kD + k] : (_Float16)0.f;
    }
#pragma unroll
    for (int n = 0; n < 7; ++n) {
      v8f c = wmma16(a, bw1[n], vzero8());
#pragma unroll
      for (int e = 0; e < 8; ++e) {
        int m   = d_m(e, lane);
        int col = n * 16 + (lane & 15);
        float t = c[e] + ((col < kH) ? lb1[col] : 0.f);
        lt[wv][m * 112 + col] = (_Float16)fmaxf(t, 0.f);
      }
    }
    __syncthreads();  // uniform trip count; orders lt write -> lt read

    v8f c2[2];
    c2[0] = vzero8();
    c2[1] = vzero8();
#pragma unroll
    for (int kk = 0; kk < 4; ++kk) {
      v16h a2;
      const int arow = lane & 15;
#pragma unroll
      for (int e = 0; e < 16; ++e) {
        int k = kk * 32 + a_k(e, lane);
        a2[e] = (k < 112) ? lt[wv][arow * 112 + k] : (_Float16)0.f;
      }
      c2[0] = wmma16(a2, bw2[kk][0], c2[0]);
      c2[1] = wmma16(a2, bw2[kk][1], c2[1]);
    }
#pragma unroll
    for (int nt = 0; nt < 2; ++nt)
#pragma unroll
      for (int e = 0; e < 8; ++e) {
        int n = nt * 16 + (lane & 15);
        if (n < kD) {
          int orow = wv * 144 + mt * 16 + d_m(e, lane);
          int osl  = orow / kS, os = orow % kS;
          float t2 = fmaxf(c2[nt][e] + lb2[n], 0.f);
          lhn[osl * kSD + os * kD + n] += t2;   // residual (unique writer)
        }
      }
    __syncthreads();
  }

  if (tid < 32) tdm_store_1d(lhn, hbuf + s0 * kSD, kSPB * kSD);  // DMA back
  for (int f = tid; f < kSD; f += 128) {
    float s1 = 0.f, s2 = 0.f;
    for (int sl = 0; sl < kSPB; ++sl) {
      float v = lhn[sl * kSD + f];
      s1 += v;
      s2 += v * v;
    }
    pS[(long)blockIdx.x * kSD + f] = s1;
    pQ[(long)blockIdx.x * kSD + f] = s2;
  }
}

// ---------------------------------------------------------------------------
// Head: feat = relu(BN7(h) * final_w^T + fb) via WMMA (K=180 pad 192,
// N=100 pad 112); then scalar 100->9 softmax and 100->1 tanh heads.
// Output: dist flat [B*9], then ev [B].
// ---------------------------------------------------------------------------
__global__ __launch_bounds__(128) void k_head(const float* __restrict__ hbuf,
                                              const float* __restrict__ fw,
                                              const float* __restrict__ fb,
                                              const float* __restrict__ dw,
                                              const float* __restrict__ db,
                                              const float* __restrict__ evw,
                                              const float* __restrict__ evb,
                                              const float* __restrict__ bnsc,
                                              float* __restrict__ out) {
  __shared__ float    lraw[kSPB * kSD]; // TDM tile
  __shared__ _Float16 lh[kSPB * kSD];   // post-BN h (f16)
  __shared__ _Float16 lfw[kH * kSD];    // final_w (f16), 36 KB
  __shared__ float    lfeat[kSPB * kH];

  const int  tid = threadIdx.x;
  const long s0  = (long)blockIdx.x * kSPB;

  if (tid < 32) {
    tdm_load_1d(lraw, hbuf + s0 * kSD, kSPB * kSD);
    __builtin_amdgcn_s_wait_tensorcnt(0);
  }
  for (int t = tid; t < kH * kSD; t += 128) lfw[t] = (_Float16)fw[t];
  __syncthreads();
  for (int t = tid; t < kSPB * kSD; t += 128) {
    int f = t % kSD;
    lh[t] = (_Float16)(lraw[t] * bnsc[f] + bnsc[kSD + f]);
  }
  __syncthreads();

  const int lane = tid & 31, wv = tid >> 5;
  const int arow = wv * 16 + (lane & 15);     // sample row
  for (int n = 0; n < 7; ++n) {               // N-tiles over 100 (pad 112)
    v8f c = vzero8();
    for (int kk = 0; kk < 6; ++kk) {          // K over 180 (pad 192)
      v16h a, b;
#pragma unroll
      for (int e = 0; e < 16; ++e) {
        int k = kk * 32 + a_k(e, lane);
        a[e] = (k < kSD) ? lh[arow * kSD + k] : (_Float16)0.f;
      }
#pragma unroll
      for (int e = 0; e < 16; ++e) {
        int kb   = kk * 32 + b_k(e, lane);
        int ncol = n * 16 + (lane & 15);
        b[e] = (kb < kSD && ncol < kH) ? lfw[ncol * kSD + kb] : (_Float16)0.f;
      }
      c = wmma16(a, b, c);
    }
#pragma unroll
    for (int e = 0; e < 8; ++e) {
      int m    = d_m(e, lane);
      int ncol = n * 16 + (lane & 15);
      if (ncol < kH)
        lfeat[(wv * 16 + m) * kH + ncol] = fmaxf(c[e] + fb[ncol], 0.f);
    }
  }
  __syncthreads();

  if (tid < kSPB) {                           // dist head + softmax
    const int sl = tid;
    float sc[kS];
    float mx = -1e30f;
    for (int o = 0; o < kS; ++o) {
      float acc = db[o];
      for (int k = 0; k < kH; ++k) acc += lfeat[sl * kH + k] * dw[o * kH + k];
      sc[o] = acc;
      mx = fmaxf(mx, acc);
    }
    float sum = 0.f;
    for (int o = 0; o < kS; ++o) {
      sc[o] = __expf(sc[o] - mx);
      sum += sc[o];
    }
    const float inv = 1.f / sum;
    for (int o = 0; o < kS; ++o) out[(s0 + sl) * kS + o] = sc[o] * inv;
  } else {                                    // eval head + tanh
    const int sl = tid - kSPB;
    float acc = evb[0];
    for (int k = 0; k < kH; ++k) acc += lfeat[sl * kH + k] * evw[k];
    out[(long)kB * kS + s0 + sl] = tanhf(acc);
  }
}

// ---------------------------------------------------------------------------
extern "C" void kernel_launch(void* const* d_in, const int* in_sizes, int n_in,
                              void* d_out, int out_size, void* d_ws,
                              size_t ws_size, hipStream_t stream) {
  (void)in_sizes; (void)n_in; (void)out_size; (void)ws_size;
  const float* x   = (const float*)d_in[0];
  const float* ew  = (const float*)d_in[1];
  const float* aw  = (const float*)d_in[2];
  const float* ab  = (const float*)d_in[3];
  const float* w1  = (const float*)d_in[4];
  const float* b1  = (const float*)d_in[5];
  const float* w2  = (const float*)d_in[6];
  const float* b2  = (const float*)d_in[7];
  const float* gam = (const float*)d_in[8];
  const float* bet = (const float*)d_in[9];
  const float* fw  = (const float*)d_in[10];
  const float* fb  = (const float*)d_in[11];
  const float* dw  = (const float*)d_in[12];
  const float* db  = (const float*)d_in[13];
  const float* evw = (const float*)d_in[14];
  const float* evb = (const float*)d_in[15];
  float*       out = (float*)d_out;

  // Workspace layout (floats): hbuf[B*180] | pS[2048*180] | pQ[2048*180] |
  // bnsc[8 stages * 2 * 180].  ~97.4 MB total; hbuf stays L2-resident.
  float* hbuf = (float*)d_ws;
  float* pS   = hbuf + (size_t)kB * kSD;
  float* pQ   = pS + (size_t)kNBLK * kSD;
  float* bnsc = pQ + (size_t)kNBLK * kSD;

  k_embed<<<(kB * kS) / 128, 256, 0, stream>>>(x, ew, hbuf);
  for (int i = 0; i < 4; ++i) {
    const float* prev = (i == 0) ? nullptr : bnsc + (size_t)(2 * i - 1) * 2 * kSD;
    k_attn<<<kNBLK, 128, 0, stream>>>(hbuf, aw + (size_t)i * 3 * kD * kD,
                                      ab + (size_t)i * 3 * kD, prev, pS, pQ);
    k_red<<<1, 192, 0, stream>>>(pS, pQ, gam + (size_t)(2 * i) * kSD,
                                 bet + (size_t)(2 * i) * kSD,
                                 bnsc + (size_t)(2 * i) * 2 * kSD, kNBLK);
    k_mlp<<<kNBLK, 128, 0, stream>>>(hbuf, w1 + (size_t)i * kH * kD,
                                     b1 + (size_t)i * kH,
                                     w2 + (size_t)i * kD * kH,
                                     b2 + (size_t)i * kD,
                                     bnsc + (size_t)(2 * i) * 2 * kSD, pS, pQ);
    k_red<<<1, 192, 0, stream>>>(pS, pQ, gam + (size_t)(2 * i + 1) * kSD,
                                 bet + (size_t)(2 * i + 1) * kSD,
                                 bnsc + (size_t)(2 * i + 1) * 2 * kSD, kNBLK);
  }
  k_head<<<kNBLK, 128, 0, stream>>>(hbuf, fw, fb, dw, db, evw, evb,
                                    bnsc + (size_t)7 * 2 * kSD, out);
}
